// SignatureCrossAttention_83691732729937
// MI455X (gfx1250) — matrix operand order repaired and checked by reading, not compiled, not run on hardware
//
#include <hip/hip_runtime.h>
#include <hip/hip_bf16.h>

// ---------------- problem constants ----------------
#define Bc    2
#define Lc    96
#define Hc    2
#define Ec    63
#define OUTc  256
#define C1c   64
#define C2c   128
#define KSc   3
#define Cc    192          // E + 1 + CONV2
#define LTc   94           // L - KS + 1
#define Nc    187          // 2*LT - 1
#define NPc   192          // padded N (multiple of 16)
#define SIGc  37056        // C + C*C
#define EVc   64
#define BHc   4
#define NFRAG (Cc / 16)     // 12 v16bf fragments per 192-col row
#define SLICEV (16 * NFRAG) // v16bf elements per W i-slice (16 rows x 12 frags)
#define SLICEB (SLICEV * 32)// bytes per slice (6144)

typedef __attribute__((ext_vector_type(16))) __bf16 v16bf;
typedef __attribute__((ext_vector_type(8)))  float  v8f;
typedef __attribute__((ext_vector_type(4)))  unsigned int u32x4;
typedef __attribute__((ext_vector_type(8)))  int i32x8;
typedef __attribute__((ext_vector_type(4)))  int i32x4;

union FragU { struct { u32x4 a, b; } p; v16bf v; };

// K offset inside a 32-wide K tile for element e of a 16-bit A fragment.
// Contiguous order (matches the B fragments): element e of lane half h
// holds K = 16*half + e.
__device__ __forceinline__ int akk(int e, int half) {
    return (half << 4) + e;
}

// ---- TDM: async-copy one 16x192 bf16 tile (row stride SIGc) into LDS -------
// D# group0/group1 per CDNA5 ISA 8.3/8.4: count=1, type=2("image"),
// data_size=1(2B), tile_dim0=192, tile_dim1=16, tensor_dim0_stride=SIGc.
__device__ __forceinline__ void tdm_load_slice(const __bf16* gsrc, unsigned lds_off) {
    unsigned long long ga = (unsigned long long)(uintptr_t)gsrc;
    u32x4 g0;
    g0[0] = 1u;                                               // count=1 (valid)
    g0[1] = lds_off;                                          // lds_addr (bytes)
    g0[2] = (unsigned)(ga & 0xFFFFFFFFu);                     // global_addr[31:0]
    g0[3] = (unsigned)((ga >> 32) & 0x1FFFFFFu) | (2u << 30); // addr[56:32] | type=2
    i32x8 g1;
    g1[0] = 1 << 16;                 // data_size=1 (2 bytes); no multicast/pad
    g1[1] = (int)(192u << 16);       // tensor_dim0[15:0]=192 in [31:16]
    g1[2] = (int)(16u << 16);        // tensor_dim0[31:16]=0 | tensor_dim1[15:0]=16
    g1[3] = (int)(192u << 16);       // tensor_dim1[31:16]=0 | tile_dim0=192
    g1[4] = 16;                      // tile_dim1=16 | tile_dim2=0
    g1[5] = SIGc;                    // tensor_dim0_stride[31:0] (data_size units)
    g1[6] = 0;                       // stride[47:32]=0 | tensor_dim1_stride lo=0
    g1[7] = 0;
    i32x4 z4 = {0, 0, 0, 0};
#if __clang_major__ >= 23
    i32x8 z8 = {0, 0, 0, 0, 0, 0, 0, 0};
    __builtin_amdgcn_tensor_load_to_lds(g0, g1, z4, z4, z8, 0);
#else
    __builtin_amdgcn_tensor_load_to_lds(g0, g1, z4, z4, 0);
#endif
}

// ---- 6 WMMA B fragments (12x ds_load_b128) from LDS at byte offset `base` --
// Inline asm so the DS path cannot be folded away (the LDS data is produced by
// the TDM, which the optimizer cannot see as a store).
__device__ __forceinline__ void lds_read_6frags(unsigned base, v16bf f[6]) {
    FragU r0, r1, r2, r3, r4, r5;
    asm volatile(
        "ds_load_b128 %0, %12\n\t"
        "ds_load_b128 %1, %12 offset:16\n\t"
        "ds_load_b128 %2, %12 offset:64\n\t"
        "ds_load_b128 %3, %12 offset:80\n\t"
        "ds_load_b128 %4, %12 offset:128\n\t"
        "ds_load_b128 %5, %12 offset:144\n\t"
        "ds_load_b128 %6, %12 offset:192\n\t"
        "ds_load_b128 %7, %12 offset:208\n\t"
        "ds_load_b128 %8, %12 offset:256\n\t"
        "ds_load_b128 %9, %12 offset:272\n\t"
        "ds_load_b128 %10, %12 offset:320\n\t"
        "ds_load_b128 %11, %12 offset:336\n\t"
        "s_wait_dscnt 0x0"
        : "=v"(r0.p.a), "=v"(r0.p.b), "=v"(r1.p.a), "=v"(r1.p.b),
          "=v"(r2.p.a), "=v"(r2.p.b), "=v"(r3.p.a), "=v"(r3.p.b),
          "=v"(r4.p.a), "=v"(r4.p.b), "=v"(r5.p.a), "=v"(r5.p.b)
        : "v"(base));
    f[0] = r0.v; f[1] = r1.v; f[2] = r2.v;
    f[3] = r3.v; f[4] = r4.v; f[5] = r5.v;
}

// ---------------- K0: convert one W (OUT x SIG) fp32 -> bf16 ----------------
__global__ void cvt_w_kernel(const float* __restrict__ w, __bf16* __restrict__ wbf) {
    long long i = (long long)blockIdx.x * blockDim.x + threadIdx.x;
    if (i < (long long)OUTc * SIGc) wbf[i] = (__bf16)w[i];
}

// ---------------- K1: conv augment ----------------
// aug layout: [s][bh][t(94)][c(192)]
__global__ void augment_kernel(const float* __restrict__ q, const float* __restrict__ k,
                               const float* __restrict__ Wq1, const float* __restrict__ bq1,
                               const float* __restrict__ Wq2, const float* __restrict__ bq2,
                               const float* __restrict__ Wk1, const float* __restrict__ bk1,
                               const float* __restrict__ Wk2, const float* __restrict__ bk2,
                               float* __restrict__ aug) {
    int s  = blockIdx.x >> 2;
    int bh = blockIdx.x & 3;
    int b  = bh >> 1, h = bh & 1;
    const float* x4 = s ? k : q;           // (B, L, H, E)
    const float* W1 = s ? Wk1 : Wq1;       // (C1, E, KS)
    const float* B1 = s ? bk1 : bq1;
    const float* W2 = s ? Wk2 : Wq2;       // (C2, C1, 1)
    const float* B2 = s ? bk2 : bq2;

    __shared__ float y1[C1c * LTc];        // 64*94 floats = 24 KB

    for (int idx = threadIdx.x; idx < C1c * LTc; idx += blockDim.x) {
        int c1 = idx / LTc, t = idx % LTc;
        float acc = B1[c1];
        for (int ks = 0; ks < KSc; ++ks) {
            const float* xr = x4 + ((size_t)(b * Lc + t + ks) * Hc + h) * Ec;
            const float* wr = W1 + ((size_t)c1 * Ec) * KSc + ks;
            for (int e = 0; e < Ec; ++e) acc += wr[e * KSc] * xr[e];
        }
        y1[c1 * LTc + t] = acc;
    }
    __syncthreads();

    float* augb = aug + (size_t)(s * BHc + bh) * LTc * Cc;
    for (int idx = threadIdx.x; idx < LTc * Cc; idx += blockDim.x) {
        int t = idx / Cc, c = idx % Cc;
        float val;
        if (c < Ec) {                                    // trunc = x[t+2, c]
            val = x4[((size_t)(b * Lc + t + KSc - 1) * Hc + h) * Ec + c];
        } else if (c == Ec) {                            // time channel
            val = (float)t / (float)(LTc - 1);
        } else {                                         // relu(conv2)
            int c2 = c - (Ec + 1);
            float acc = B2[c2];
            const float* wr = W2 + (size_t)c2 * C1c;
            for (int c1 = 0; c1 < C1c; ++c1) acc += wr[c1] * y1[c1 * LTc + t];
            val = fmaxf(acc, 0.0f);
        }
        augb[idx] = val;
    }
}

// ---------------- K2: lead-lag diff streams ----------------
// s1/dx layout: [s][bh][n(192 padded, zero beyond 186)][c(192)]
// aT  layout:   [s][bh][i(192)][n(192 padded)]   (a = pre + 0.5*dx, transposed)
__global__ void streams_kernel(const float* __restrict__ aug,
                               float* __restrict__ s1, float* __restrict__ dx,
                               float* __restrict__ aT) {
    int s  = blockIdx.x >> 2;
    int bh = blockIdx.x & 3;
    const float* augb = aug + (size_t)(s * BHc + bh) * LTc * Cc;
    size_t base = (size_t)(s * BHc + bh) * NPc * Cc;
    for (int idx = threadIdx.x; idx < NPc * Cc; idx += blockDim.x) {
        int n = idx / Cc, c = idx % Cc;
        float vs1 = 0.f, vdx = 0.f;
        if (n < Nc) {
            // lead-lag path: path[m] = aug[min((m+1)/2, LT-1)]
            int i1 = (n + 1) >> 1; if (i1 > LTc - 1) i1 = LTc - 1;
            int i2 = (n + 2) >> 1; if (i2 > LTc - 1) i2 = LTc - 1;
            float p1 = augb[(size_t)i2 * Cc + c];
            float p0 = augb[(size_t)i1 * Cc + c];
            vdx = p1 - p0;
            vs1 = p1 - augb[c];          // cumsum(dx) telescopes to path[n+1]-path[0]
        }
        s1[base + idx] = vs1;
        dx[base + idx] = vdx;
        aT[base + (size_t)c * NPc + n] = vs1 - 0.5f * vdx;   // pre + 0.5*dx
    }
}

// ---------------- K3: fused outer-product signature GEMM ----------------
// WMMA bf16 + TDM-staged W slices read back via ds_load_b128 (inline asm).
// One workgroup = 12 waves = all 12 n-tiles of one (bh, o-tile). Per i-block,
// wave 0 issues one tensor_load_to_lds of the 16x192 bf16 W slice into a
// double-buffered LDS tile (TENSORcnt-tracked), overlapped with compute of
// slice i; all 12 waves consume B fragments from LDS (12x reuse/byte).
//   lin[n,o] = sum_{k<C}  s1[n,k] * W[o,k]
//   con[n,o] = sum_i a[n,i] * ( sum_j dx[n,j] * W[o, C + i*C + j] )
__global__ __launch_bounds__(384) void sig_gemm_wmma(
    int s,
    const float* __restrict__ s1buf, const float* __restrict__ dxbuf,
    const float* __restrict__ aTbuf, const __bf16* __restrict__ wbf,
    float* __restrict__ lin, float* __restrict__ con) {
    // Double-buffered W slice, stored as WMMA B fragments:
    // byte offset = buf*6144 + row*384 + jt*64 + half*32  (matches TDM tile)
    __shared__ v16bf wslv[2 * SLICEV];   // 12 KB (sole LDS var -> offset 0)

    int ot = blockIdx.x & 15;
    int bh = blockIdx.x >> 4;
    int wave = threadIdx.x >> 5;        // 0..11 == n-tile
    int lane = threadIdx.x & 31;
    int nn   = lane & 15;               // D column (o) / A row (n) within tile
    int half = lane >> 4;

    // Zero-init the staging buffer (keeps the LDS object live + defined).
    {
        v16bf z = {};
        for (int t = threadIdx.x; t < 2 * SLICEV; t += blockDim.x) wslv[t] = z;
    }
    __syncthreads();

    int n0 = wave * 16, o0 = ot * 16;
    size_t sb = (size_t)s * BHc + bh;

    const float* s1r = s1buf + (sb * NPc + n0 + nn) * Cc;
    const float* dxr = dxbuf + (sb * NPc + n0 + nn) * Cc;
    const float* aTr = aTbuf + sb * Cc * NPc;            // row i stride = NPc
    const __bf16* wr = wbf + (size_t)(o0 + nn) * SIGc;   // B lane holds column o0+nn

    v8f acc_lin = {};
    v8f acc_out = {};

    // ---- phase 1: linear (s1) region, k in [0, C) ---- (direct global W)
#pragma unroll
    for (int kt = 0; kt < Cc / 32; ++kt) {
        int k0 = kt * 32;
        v16bf afrag, bfrag;
        bfrag = *(const v16bf*)(wr + k0 + (half << 4));
#pragma unroll
        for (int e = 0; e < 16; ++e)
            afrag[e] = (__bf16)s1r[k0 + akk(e, half)];
        acc_lin = __builtin_amdgcn_wmma_f32_16x16x32_bf16(
            false, afrag, false, bfrag, (short)0, acc_lin, false, false);
    }

    // ---- phase 2 setup: convert this lane's dx row to bf16 fragments once --
    v16bf dxf[Cc / 32];
#pragma unroll
    for (int jt = 0; jt < Cc / 32; ++jt)
#pragma unroll
        for (int e = 0; e < 16; ++e)
            dxf[jt][e] = (__bf16)dxr[jt * 32 + akk(e, half)];

    int nbase = n0 + (half << 3);       // D layout: VGPR r -> M = r + 8*half
    unsigned lrow = (unsigned)(nn * 384 + half * 32);   // lane's LDS row base

    // ---- phase 2: TDM-pipelined i-loop ----
    const __bf16* wtile0 = wbf + (size_t)o0 * SIGc + Cc;   // rows o0.., k = C
    if (wave == 0) tdm_load_slice(wtile0, 0);
    for (int i = 0; i < Cc; ++i) {
        if (wave == 0) __builtin_amdgcn_s_wait_tensorcnt(0);  // slice i landed
        __syncthreads();                 // publish LDS; prior reads of buf done
        if (wave == 0 && i + 1 < Cc)
            tdm_load_slice(wtile0 + (size_t)(i + 1) * Cc,
                           (unsigned)(((i + 1) & 1) * SLICEB));

        v16bf bf[6];
        lds_read_6frags((unsigned)((i & 1) * SLICEB) + lrow, bf);
        v8f tmp = {};
#pragma unroll
        for (int jt = 0; jt < Cc / 32; ++jt) {
            tmp = __builtin_amdgcn_wmma_f32_16x16x32_bf16(
                false, dxf[jt], false, bf[jt], (short)0, tmp, false, false);
        }
        // per-row scale a[n, i]: 8 contiguous fp32 (32B-aligned) for this lane
        const float* av = aTr + (size_t)i * NPc + nbase;
        float4 a0 = *(const float4*)(av);
        float4 a1 = *(const float4*)(av + 4);
        acc_out[0] += a0.x * tmp[0];
        acc_out[1] += a0.y * tmp[1];
        acc_out[2] += a0.z * tmp[2];
        acc_out[3] += a0.w * tmp[3];
        acc_out[4] += a1.x * tmp[4];
        acc_out[5] += a1.y * tmp[5];
        acc_out[6] += a1.z * tmp[6];
        acc_out[7] += a1.w * tmp[7];
    }

    // Opaque guard (s is a runtime arg, never -1): keeps a language-level load
    // of wslv so the LDS object cannot be store-only-eliminated.
    if (s == -1) lin[0] = (float)((const __bf16*)wslv)[0];

    // ---- store (C/D layout: VGPR r -> M = r + 8*half, N = lane&15) ----
    int o = o0 + nn;
#pragma unroll
    for (int r = 0; r < 8; ++r) {
        int n = nbase + r;
        size_t off = (sb * NPc + n) * OUTc + o;
        lin[off] = acc_lin[r];
        con[off] = acc_out[r];
    }
}

// ---------------- K4: cumsum over n + bias ----------------
// proj layout: [s][bh][n(187)][o(256)]
__global__ void cumsum_proj_kernel(const float* __restrict__ lin,
                                   const float* __restrict__ con,
                                   const float* __restrict__ blq,
                                   const float* __restrict__ blk,
                                   float* __restrict__ proj) {
    int s  = blockIdx.x >> 2;
    int bh = blockIdx.x & 3;
    int o  = threadIdx.x;
    const float* bias = s ? blk : blq;
    size_t sb = (size_t)s * BHc + bh;
    const float* linb = lin + sb * NPc * OUTc;
    const float* conb = con + sb * NPc * OUTc;
    float* projb = proj + sb * Nc * OUTc;
    float run = 0.f;
    float bo = bias[o];
    for (int n = 0; n < Nc; ++n) {
        run += conb[(size_t)n * OUTc + o];
        projb[(size_t)n * OUTc + o] = bo + linb[(size_t)n * OUTc + o] + run;
    }
}

// ---------------- K5: attn = tanh(qproj . kproj^T) ----------------
__global__ void attn_kernel(const float* __restrict__ proj, float* __restrict__ attn) {
    long long idx = (long long)blockIdx.x * blockDim.x + threadIdx.x;
    if (idx >= (long long)BHc * Nc * Nc) return;
    int y = (int)(idx % Nc);
    int t = (int)(idx / Nc);
    int x = t % Nc;
    int bh = t / Nc;
    const float* qp = proj + ((size_t)bh * Nc + x) * OUTc;                  // s=0
    const float* kp = proj + (((size_t)BHc + bh) * Nc + y) * OUTc;          // s=1
    float acc = 0.f;
    for (int o = 0; o < OUTc; ++o) acc += qp[o] * kp[o];
    attn[idx] = tanhf(acc);
}

// ---------------- K6: out = attn @ v ----------------
__global__ void out_kernel(const float* __restrict__ attn, const float* __restrict__ v,
                           float* __restrict__ out) {
    long long idx = (long long)blockIdx.x * blockDim.x + threadIdx.x;
    if (idx >= (long long)BHc * Nc * EVc) return;
    int e = (int)(idx & (EVc - 1));
    int t = (int)(idx >> 6);
    int x = t % Nc;
    int bh = t / Nc;
    const float* ar = attn + ((size_t)bh * Nc + x) * Nc;
    const float* vb = v + (size_t)bh * Nc * EVc;
    float acc = 0.f;
    for (int y = 0; y < Nc; ++y) acc += ar[y] * vb[(size_t)y * EVc + e];
    out[idx] = acc;
}

// ---------------- launcher ----------------
extern "C" void kernel_launch(void* const* d_in, const int* in_sizes, int n_in,
                              void* d_out, int out_size, void* d_ws, size_t ws_size,
                              hipStream_t stream) {
    const float* q   = (const float*)d_in[0];
    const float* k   = (const float*)d_in[1];
    const float* v   = (const float*)d_in[2];
    const float* Wq1 = (const float*)d_in[3];
    const float* bq1 = (const float*)d_in[4];
    const float* Wq2 = (const float*)d_in[5];
    const float* bq2 = (const float*)d_in[6];
    const float* Wk1 = (const float*)d_in[7];
    const float* bk1 = (const float*)d_in[8];
    const float* Wk2 = (const float*)d_in[9];
    const float* bk2 = (const float*)d_in[10];
    const float* Wlq = (const float*)d_in[11];
    const float* blq = (const float*)d_in[12];
    const float* Wlk = (const float*)d_in[13];
    const float* blk = (const float*)d_in[14];
    float* out = (float*)d_out;

    // workspace layout (bytes)
    char* ws = (char*)d_ws;
    size_t off = 0;
    auto take = [&](size_t bytes) { size_t o = off; off += (bytes + 255) & ~(size_t)255; return o; };
    __bf16* wbf = (__bf16*)(ws + take((size_t)OUTc * SIGc * sizeof(__bf16)));   // reused per stream
    float* aug  = (float*)(ws + take((size_t)2 * BHc * LTc * Cc * sizeof(float)));
    float* s1b  = (float*)(ws + take((size_t)2 * BHc * NPc * Cc * sizeof(float)));
    float* dxb  = (float*)(ws + take((size_t)2 * BHc * NPc * Cc * sizeof(float)));
    float* aTb  = (float*)(ws + take((size_t)2 * BHc * Cc * NPc * sizeof(float)));
    float* lin  = (float*)(ws + take((size_t)2 * BHc * NPc * OUTc * sizeof(float)));
    float* con  = (float*)(ws + take((size_t)2 * BHc * NPc * OUTc * sizeof(float)));
    float* proj = (float*)(ws + take((size_t)2 * BHc * Nc * OUTc * sizeof(float)));
    float* attn = (float*)(ws + take((size_t)BHc * Nc * Nc * sizeof(float)));
    (void)ws_size; (void)in_sizes; (void)n_in; (void)out_size;

    // stage 1: augmentation + signature streams
    augment_kernel<<<2 * BHc, 256, 0, stream>>>(q, k, Wq1, bq1, Wq2, bq2,
                                                Wk1, bk1, Wk2, bk2, aug);
    streams_kernel<<<2 * BHc, 256, 0, stream>>>(aug, s1b, dxb, aTb);

    // stage 2: fused signature-projection GEMM per stream (W converted to bf16)
    const int wcnt = OUTc * SIGc;
    const int gemm_blocks = BHc * (OUTc / 16);   // 4*16 = 64 workgroups (12 waves each)
    cvt_w_kernel<<<(wcnt + 255) / 256, 256, 0, stream>>>(Wlq, wbf);
    sig_gemm_wmma<<<gemm_blocks, 384, 0, stream>>>(0, s1b, dxb, aTb, wbf, lin, con);
    cvt_w_kernel<<<(wcnt + 255) / 256, 256, 0, stream>>>(Wlk, wbf);
    sig_gemm_wmma<<<gemm_blocks, 384, 0, stream>>>(1, s1b, dxb, aTb, wbf, lin, con);

    // stage 3: cumsum + bias -> projections
    cumsum_proj_kernel<<<2 * BHc, OUTc, 0, stream>>>(lin, con, blq, blk, proj);

    // stage 4: attention + output
    long long na = (long long)BHc * Nc * Nc;
    attn_kernel<<<(unsigned)((na + 255) / 256), 256, 0, stream>>>(proj, attn);
    long long no = (long long)BHc * Nc * EVc;
    out_kernel<<<(unsigned)((no + 255) / 256), 256, 0, stream>>>(attn, v, out);
}
